// CSCTV_42451456754126
// MI455X (gfx1250) — compile-verified, matrix operand order checked by reading, not run on hardware
//
#include <hip/hip_runtime.h>
#include <math.h>

// ---------------- problem constants (from reference) ----------------
#define MBB   2
#define CCH   3
#define IMH   256
#define IMW   256
#define PCH   32
#define PIXN  (IMH*IMW)          // 65536
#define NSTEP 10
#define EPSR  1e-8f

// ---------------- vector types ----------------
typedef __bf16        v16bf __attribute__((ext_vector_type(16)));
typedef float         v8f   __attribute__((ext_vector_type(8)));
typedef unsigned int  u32x4 __attribute__((ext_vector_type(4)));
typedef unsigned int  u32x8 __attribute__((ext_vector_type(8)));
typedef int           i32x4 __attribute__((ext_vector_type(4)));
typedef int           i32x8 __attribute__((ext_vector_type(8)));

#if defined(__has_builtin)
#if __has_builtin(__builtin_amdgcn_tensor_load_to_lds)
#define HAS_TDM 1
#endif
#endif

// 5x5 depthwise correlation with zero padding (fil is symmetric in the
// reference, so conv_fil == convT_fil).
__device__ __forceinline__ float conv5(const float* __restrict__ plane,
                                       const float* __restrict__ fil,
                                       int y, int x) {
  float s = 0.f;
#pragma unroll
  for (int u = 0; u < 5; ++u) {
    int yy = y + u - 2;
    if (yy < 0 || yy >= IMH) continue;
#pragma unroll
    for (int v = 0; v < 5; ++v) {
      int xx = x + v - 2;
      if (xx < 0 || xx >= IMW) continue;
      s += plane[yy*IMW + xx] * fil[u*5 + v];
    }
  }
  return s;
}

// ---------------- K0: x0 = convT_fil(z) ----------------
__global__ void __launch_bounds__(256)
k_init_x(const float* __restrict__ z, const float* __restrict__ fil,
         float* __restrict__ x) {
  int idx = blockIdx.x*256 + threadIdx.x;
  if (idx >= MBB*CCH*PIXN) return;
  int pix = idx % PIXN, pl = idx / PIXN;
  x[idx] = conv5(z + (size_t)pl*PIXN, fil, pix/IMW, pix%IMW);
}

// ---------------- K1: Ba = conv_CSC(a, B)  (WMMA) ----------------
// Per block: one (mb,c) and one 8x8 output tile. Tap-expansion GEMM:
//   U[hpix, tap] = sum_p a_lds[hpix, p] * B[p, c, tap]   (K = 32 exactly)
// over the 14x14 halo region, then Ba[pix] = sum_{tap} U[pix + d(tap), tap].
__global__ void __launch_bounds__(256)
k_conv_csc(const float* __restrict__ a, const float* __restrict__ Bw,
           float* __restrict__ Ba) {
  __shared__ __attribute__((aligned(16))) __bf16 a_lds[208*32];   // [hpix][p]
  __shared__ __attribute__((aligned(16))) __bf16 w_lds[64*32];    // [tap][p]
  __shared__ __attribute__((aligned(16))) __bf16 u_lds[208*64];   // [hpix][tap]
  const int tid = threadIdx.x;
  int b = blockIdx.x;
  const int tile = b & 1023; b >>= 10;
  const int c  = b % CCH;
  const int mb = b / CCH;
  const int ty = (tile >> 5) * 8;
  const int tx = (tile & 31) * 8;

  // stage a halo tile (14x14, 3-pixel halo around 8x8) for all 32 p, as bf16
  for (int p = 0; p < PCH; ++p) {
    const float* src = a + ((size_t)(mb*PCH + p)*CCH + c)*PIXN;
    if (p + 1 < PCH) __builtin_prefetch(src + CCH*PIXN, 0, 0);
    if (tid < 208) {
      float v = 0.f;
      if (tid < 196) {
        int hy = tid / 14, hx = tid % 14;
        int gy = ty - 3 + hy, gx = tx - 3 + hx;
        if (gy >= 0 && gy < IMH && gx >= 0 && gx < IMW) v = src[gy*IMW + gx];
      }
      a_lds[tid*32 + p] = (__bf16)v;
    }
  }
  // weights: w_lds[t][p] = B[p,c,t], taps padded 49 -> 64 with zeros
  for (int i = tid; i < 64*32; i += 256) {
    int t = i >> 5, p = i & 31;
    float v = (t < 49) ? Bw[((size_t)p*CCH + c)*49 + t] : 0.f;
    w_lds[t*32 + p] = (__bf16)v;
  }
  __syncthreads();

  const int wave = tid >> 5, lane = tid & 31;
  const int lrow = lane & 15, lhi = lane >> 4, kb = lhi*8;
  // 13 M-tiles (208 halo pixels) x 4 N-tiles (64 taps) = 52 WMMA jobs / 8 waves
  for (int job = wave; job < 52; job += 8) {
    const int mtile = job >> 2, ntile = job & 3;
    const int pixr = mtile*16 + lrow;
    const int tcol = ntile*16 + lrow;
    u32x8 aw, bw;
#pragma unroll
    for (int q = 0; q < 4; ++q) {
      aw[q]   = *(const unsigned int*)&a_lds[pixr*32 + kb + 2*q];
      aw[q+4] = *(const unsigned int*)&a_lds[pixr*32 + 16 + kb + 2*q];
      bw[q]   = *(const unsigned int*)&w_lds[tcol*32 + kb + 2*q];
      bw[q+4] = *(const unsigned int*)&w_lds[tcol*32 + 16 + kb + 2*q];
    }
    v8f acc = {};
    acc = __builtin_amdgcn_wmma_f32_16x16x32_bf16(
        false, __builtin_bit_cast(v16bf, aw),
        false, __builtin_bit_cast(v16bf, bw),
        (short)0, acc, false, false);
#pragma unroll
    for (int q = 0; q < 8; ++q) {
      int m = mtile*16 + q + 8*lhi;
      u_lds[m*64 + tcol] = (__bf16)acc[q];
    }
  }
  __syncthreads();

  // gather the 49 shifted tap columns into the 8x8 output tile
  if (tid < 64) {
    int oy = tid >> 3, ox = tid & 7;
    float s = 0.f;
#pragma unroll
    for (int u = 0; u < 7; ++u)
#pragma unroll
      for (int v = 0; v < 7; ++v)
        s += (float)u_lds[((oy+u)*14 + (ox+v))*64 + (u*7 + v)];
    Ba[((size_t)mb*CCH + c)*PIXN + (size_t)(ty+oy)*IMW + (tx+ox)] = s;
  }
}

// ---------------- K2: x update (+ v = 2x_aft - x_bef) ----------------
__global__ void __launch_bounds__(256)
k_x_update(const float* __restrict__ xb, const float* __restrict__ Bacc,
           const float* __restrict__ y1, const float* __restrict__ y2,
           const float* __restrict__ fil, const float* __restrict__ gam1,
           int step, float* __restrict__ xa, float* __restrict__ vb) {
  int idx = blockIdx.x*256 + threadIdx.x;
  if (idx >= MBB*CCH*PIXN) return;
  const float g1 = gam1[step];
  int pix = idx % PIXN, pl = idx / PIXN;
  int mb = pl / CCH, c = pl % CCH;
  int y = pix / IMW, x = pix % IMW;
  int ym1 = y ? y-1 : IMH-1;
  int xm1 = x ? x-1 : IMW-1;
  const float* y1a = y1 + (((size_t)mb*2 + 0)*CCH + c)*PIXN;
  const float* y1b = y1 + (((size_t)mb*2 + 1)*CCH + c)*PIXN;
  float dt = y1a[ym1*IMW + x] - y1a[pix] + y1b[y*IMW + xm1] - y1b[pix];
  float cf = conv5(y2 + (size_t)pl*PIXN, fil, y, x);
  float xv = xb[idx];
  float nx = xv - g1*(xv - Bacc[idx] + dt + cf);
  nx = fminf(fmaxf(nx, 0.f), 1.f);
  xa[idx] = nx;
  vb[idx] = 2.f*nx - xv;
}

// ---------------- K3: a = soft(a + g2*convT_CSC(x-Ba, B), g2*l1) (WMMA) ----
// Per block: one (mb,c) and one 16x16 pixel tile.
//   D[p, pix] = W[p, tap] x Patch[tap, pix];  K = 49 -> 64 (two bf16 chunks)
// Weight slice B[:,c,:,:] is pulled into LDS by the Tensor Data Mover.
__global__ void __launch_bounds__(256)
k_convt_csc(const float* __restrict__ xb, const float* __restrict__ Bacc,
            const float* __restrict__ Bw, float* __restrict__ a,
            const float* __restrict__ lam1, const float* __restrict__ gam2,
            int step) {
  __shared__ __attribute__((aligned(16))) float  r_lds[22*22];
  __shared__ __attribute__((aligned(16))) float  wstage[32*49];
  __shared__ __attribute__((aligned(16))) __bf16 wT_lds[32*64];   // [p][tap]
  __shared__ __attribute__((aligned(16))) __bf16 p_lds[256*64];   // [pix][tap]
  const int tid = threadIdx.x;
  int b = blockIdx.x;
  const int tile = b & 255; b >>= 8;
  const int c  = b % CCH;
  const int mb = b / CCH;
  const int ty = (tile >> 4) * 16, tx = (tile & 15) * 16;
  const float g2 = gam2[step], l1 = lam1[step];

  // --- TDM: load B[:, c, :, :] (32 rows x 49 f32, row stride 147) into LDS
#ifdef HAS_TDM
  if (tid < 32) {  // wave 0 issues one tensor op
    unsigned int ldsoff = (unsigned int)(uintptr_t)(&wstage[0]);
    unsigned long long ga =
        (unsigned long long)(uintptr_t)(Bw + (size_t)c*49);
    u32x4 g0;
    g0[0] = 1u;                                   // count=1, user descriptor
    g0[1] = ldsoff;                               // lds_addr
    g0[2] = (unsigned int)ga;                     // global_addr lo
    g0[3] = (unsigned int)(ga >> 32) | (2u << 30);// global_addr hi | type=2
    i32x8 g1d;
    g1d[0] = (int)(2u << 16);                     // data_size = 4B, mask=0
    g1d[1] = (int)(49u << 16);                    // tensor_dim0 = 49
    g1d[2] = (int)(32u << 16);                    // tensor_dim1 = 32
    g1d[3] = (int)(49u << 16);                    // tile_dim0 = 49
    g1d[4] = 32;                                  // tile_dim1 = 32
    g1d[5] = 147;                                 // tensor_dim0_stride
    g1d[6] = 0;
    g1d[7] = 0;
    i32x4 gz  = {0, 0, 0, 0};
    i32x8 gz8 = {0, 0, 0, 0, 0, 0, 0, 0};
    __builtin_amdgcn_tensor_load_to_lds(g0, g1d, gz, gz, gz8, 0);
    __builtin_amdgcn_s_wait_tensorcnt(0);
  }
#else
  if (tid < 32)
    for (int i = tid; i < 32*49; i += 32)
      wstage[i] = Bw[(i/49)*CCH*49 + c*49 + (i%49)];
#endif

  // --- residual halo tile r = x_bef - Ba (22x22, zero padded)
  const float* xp = xb   + ((size_t)mb*CCH + c)*PIXN;
  const float* bp = Bacc + ((size_t)mb*CCH + c)*PIXN;
  for (int i = tid; i < 484; i += 256) {
    int hy = i / 22, hx = i % 22;
    int gy = ty - 3 + hy, gx = tx - 3 + hx;
    float v = 0.f;
    if (gy >= 0 && gy < IMH && gx >= 0 && gx < IMW) {
      int g = gy*IMW + gx;
      v = xp[g] - bp[g];
    }
    r_lds[i] = v;
  }
  __syncthreads();

  // --- bf16 weight matrix (A operand), taps padded to 64
  for (int i = tid; i < 32*64; i += 256) {
    int t = i & 63;
    wT_lds[i] = (__bf16)((t < 49) ? wstage[(i >> 6)*49 + t] : 0.f);
  }
  // --- im2col patches (B operand): pixel per thread, reversed-tap offsets
  {
    int py = tid >> 4, px = tid & 15;
#pragma unroll
    for (int t = 0; t < 64; ++t) {
      float v = 0.f;
      if (t < 49) {
        int u = t / 7, vv = t % 7;
        v = r_lds[(py + 6 - u)*22 + (px + 6 - vv)];
      }
      p_lds[tid*64 + t] = (__bf16)v;
    }
  }
  __syncthreads();

  const int wave = tid >> 5, lane = tid & 31;
  const int lrow = lane & 15, lhi = lane >> 4, kb = lhi*8;
  // 2 p-tiles x 16 pixel rows = 32 WMMA jobs / 8 waves
  for (int j = wave; j < 32; j += 8) {
    const int ptile = j & 1, prow = j >> 1;
    const int arow = ptile*16 + lrow;       // A row   (output channel p)
    const int pixn = prow*16 + lrow;        // B column (pixel in tile row)
    v8f acc = {};
#pragma unroll
    for (int kc = 0; kc < 2; ++kc) {
      u32x8 aw, bw;
#pragma unroll
      for (int q = 0; q < 4; ++q) {
        int t0 = kc*32 + kb + 2*q;
        aw[q]   = *(const unsigned int*)&wT_lds[arow*64 + t0];
        aw[q+4] = *(const unsigned int*)&wT_lds[arow*64 + t0 + 16];
        bw[q]   = *(const unsigned int*)&p_lds[pixn*64 + t0];
        bw[q+4] = *(const unsigned int*)&p_lds[pixn*64 + t0 + 16];
      }
      acc = __builtin_amdgcn_wmma_f32_16x16x32_bf16(
          false, __builtin_bit_cast(v16bf, aw),
          false, __builtin_bit_cast(v16bf, bw),
          (short)0, acc, false, false);
    }
    // fused soft-threshold update of a (coalesced along pixel lanes)
    const int gy = ty + prow;
#pragma unroll
    for (int q = 0; q < 8; ++q) {
      int p = ptile*16 + q + 8*lhi;
      size_t gi = (((size_t)mb*PCH + p)*CCH + c)*PIXN
                + (size_t)gy*IMW + tx + lrow;
      float u = a[gi] + g2*acc[q];
      float m = fabsf(u) - g2*l1;
      a[gi] = (m > 0.f) ? copysignf(m, u) : 0.f;
    }
  }
}

// ---------------- K4: y1 update (D + prox_l12, periodic wrap) ----------
__global__ void __launch_bounds__(256)
k_y1_update(const float* __restrict__ vb, float* __restrict__ y1,
            const float* __restrict__ lam2, const float* __restrict__ gam3,
            int step) {
  int idx = blockIdx.x*256 + threadIdx.x;
  if (idx >= MBB*CCH*PIXN) return;
  const float g3 = gam3[step], l2 = lam2[step];
  int pix = idx % PIXN, pl = idx / PIXN;
  int y = pix / IMW, x = pix % IMW;
  const float* vp = vb + (size_t)pl*PIXN;
  float v0 = vp[pix];
  int yp1 = (y == IMH-1) ? 0 : y+1;
  int xp1 = (x == IMW-1) ? 0 : x+1;
  float d0 = vp[yp1*IMW + x] - v0;
  float d1 = vp[y*IMW + xp1] - v0;
  int mb = pl / CCH, c = pl % CCH;
  size_t i0 = (((size_t)mb*2 + 0)*CCH + c)*PIXN + pix;
  size_t i1 = (((size_t)mb*2 + 1)*CCH + c)*PIXN + pix;
  float a0 = y1[i0] + g3*d0;
  float a1 = y1[i1] + g3*d1;
  float ge = g3 + EPSR;
  float z0 = a0/ge, z1 = a1/ge;
  float nrm = sqrtf(z0*z0 + z1*z1);
  float f = fmaxf(1.f - (l2/ge)/fmaxf(nrm, EPSR), 0.f);
  y1[i0] = a0 - g3*(z0*f);
  y1[i1] = a1 - g3*(z1*f);
}

// ---------------- K5a: y2 <- y2 + g3*conv_fil(v); partial ||.||^2 -------
__global__ void __launch_bounds__(256)
k_y2_stage(const float* __restrict__ vb, float* __restrict__ y2,
           const float* __restrict__ z, const float* __restrict__ fil,
           const float* __restrict__ gam3, int step, float* __restrict__ part) {
  __shared__ float red[256];
  const int mb = blockIdx.y;
  int i = blockIdx.x*256 + threadIdx.x;        // 768*256 == C*PIX exactly
  const float g3 = gam3[step];
  int c = i / PIXN, pix = i % PIXN;
  size_t g = ((size_t)mb*CCH + c)*PIXN + pix;
  const float* vp = vb + ((size_t)mb*CCH + c)*PIXN;
  float t = y2[g] + g3*conv5(vp, fil, pix/IMW, pix%IMW);
  y2[g] = t;
  float d = t/(g3 + EPSR) - z[g];
  red[threadIdx.x] = d*d;
  __syncthreads();
  for (int off = 128; off > 0; off >>= 1) {
    if (threadIdx.x < off) red[threadIdx.x] += red[threadIdx.x + off];
    __syncthreads();
  }
  if (threadIdx.x == 0) part[mb*768 + blockIdx.x] = red[0];
}

// ---------------- K5r: deterministic second-pass reduction --------------
__global__ void __launch_bounds__(256)
k_reduce(const float* __restrict__ part, float* __restrict__ rn) {
  __shared__ float red[256];
  int mb = blockIdx.x;
  float s = 0.f;
  for (int i = threadIdx.x; i < 768; i += 256) s += part[mb*768 + i];
  red[threadIdx.x] = s;
  __syncthreads();
  for (int off = 128; off > 0; off >>= 1) {
    if (threadIdx.x < off) red[threadIdx.x] += red[threadIdx.x + off];
    __syncthreads();
  }
  if (threadIdx.x == 0) rn[mb] = sqrtf(red[0]);
}

// ---------------- K5b: y2 <- y2 - g3*proj_l2ball(y2/g3e, z, eps) --------
__global__ void __launch_bounds__(256)
k_y2_final(float* __restrict__ y2, const float* __restrict__ z,
           const float* __restrict__ rn, const float* __restrict__ gam3,
           int step, float epsball) {
  int idx = blockIdx.x*256 + threadIdx.x;
  if (idx >= MBB*CCH*PIXN) return;
  int mb = idx / (CCH*PIXN);
  float g3 = gam3[step], ge = g3 + EPSR;
  float t = y2[idx], zz = z[idx];
  float scale = fminf(1.f, epsball / fmaxf(rn[mb], EPSR));
  float proj = zz + (t/ge - zz)*scale;
  y2[idx] = t - g3*proj;
}

// ---------------- host driver ----------------
extern "C" void kernel_launch(void* const* d_in, const int* in_sizes, int n_in,
                              void* d_out, int out_size, void* d_ws, size_t ws_size,
                              hipStream_t stream) {
  (void)in_sizes; (void)n_in; (void)out_size; (void)ws_size;
  const float* z      = (const float*)d_in[0];
  const float* a_init = (const float*)d_in[1];
  const float* Bw     = (const float*)d_in[2];
  const float* fil    = (const float*)d_in[3];
  const float* lam1   = (const float*)d_in[4];
  const float* lam2   = (const float*)d_in[5];
  const float* gam1   = (const float*)d_in[6];
  const float* gam2   = (const float*)d_in[7];
  const float* gam3   = (const float*)d_in[8];
  float* out = (float*)d_out;

  const size_t NXC = (size_t)MBB*CCH*PIXN;       // 393216
  float* ws  = (float*)d_ws;
  float* x_a = ws;
  float* x_b = x_a + NXC;
  float* Ba  = x_b + NXC;
  float* y1  = Ba  + NXC;
  float* y2  = y1  + 2*NXC;
  float* vb  = y2  + NXC;
  float* aa  = vb  + NXC;
  float* prt = aa  + (size_t)MBB*PCH*CCH*PIXN;   // 12582912
  float* rn  = prt + MBB*768;

  const float epsball = 0.05f * sqrtf((float)(CCH*IMH*IMW));

  (void)hipMemcpyAsync(aa, a_init, sizeof(float)*(size_t)MBB*PCH*CCH*PIXN,
                       hipMemcpyDeviceToDevice, stream);
  (void)hipMemsetAsync(y1, 0, sizeof(float)*2*NXC, stream);
  (void)hipMemsetAsync(y2, 0, sizeof(float)*NXC, stream);

  k_init_x<<<1536, 256, 0, stream>>>(z, fil, x_a);

  float* xb = x_a;
  float* xa = x_b;
  for (int s = 0; s < NSTEP; ++s) {
    k_conv_csc <<<MBB*CCH*1024, 256, 0, stream>>>(aa, Bw, Ba);
    k_x_update <<<1536, 256, 0, stream>>>(xb, Ba, y1, y2, fil, gam1, s, xa, vb);
    k_convt_csc<<<MBB*CCH*256, 256, 0, stream>>>(xb, Ba, Bw, aa, lam1, gam2, s);
    k_y1_update<<<1536, 256, 0, stream>>>(vb, y1, lam2, gam3, s);
    dim3 g5(768, MBB);
    k_y2_stage <<<g5, 256, 0, stream>>>(vb, y2, z, fil, gam3, s, prt);
    k_reduce   <<<MBB, 256, 0, stream>>>(prt, rn);
    k_y2_final <<<1536, 256, 0, stream>>>(y2, z, rn, gam3, s, epsball);
    float* t = xb; xb = xa; xa = t;
  }
  // after 10 swaps, xb points at the final x
  (void)hipMemcpyAsync(out, xb, sizeof(float)*NXC, hipMemcpyDeviceToDevice, stream);
}
